// SensorConvLSTM_72026601554117
// MI455X (gfx1250) — matrix operand order repaired and checked by reading, not compiled, not run on hardware
//
#include <hip/hip_runtime.h>
#include <hip/hip_bf16.h>

typedef _Float16 half8  __attribute__((ext_vector_type(8)));
typedef _Float16 half16 __attribute__((ext_vector_type(16)));
typedef float    f32x8  __attribute__((ext_vector_type(8)));

#define NSENS   21
#define IN_CH   3
#define HID     128
#define T_LEN   128
#define BATCH   256
#define OC      512            // 4*HID gate channels
#define KREAL   393            // 131 * 3 taps (logical)
#define TAPSTR  136            // halves per tap block (16B aligned segments)
#define KBIAS   408            // 3*136; bias column
#define KPAD    416            // 13 K-tiles of 32
#define KTILES  13
#define NCOL    32             // padded sensor width (2 N-tiles)
#define GSTRIDE 22             // gate-buffer row stride (floats)
#define NTHREADS 256

// ---------------------------------------------------------------------------
// Kernel 1: pack conv weights (512 x 131 x 1 x 3 f32) + bias into fp16
// [OC][KPAD], tap-major: kk = tap*136 + c', c'<128 -> h-channel (ic=c'+3),
// c' in 128..130 -> x-channel (ic=c'-128), kk==408 -> bias, rest 0.
// ---------------------------------------------------------------------------
__global__ __launch_bounds__(256) void pack_weights_kernel(
    const float* __restrict__ conv_w, const float* __restrict__ conv_b,
    _Float16* __restrict__ wh)
{
    int idx = blockIdx.x * 256 + threadIdx.x;
    if (idx >= OC * KPAD) return;
    int oc = idx / KPAD;
    int kk = idx - oc * KPAD;
    float v = 0.0f;
    if (kk < KBIAS) {
        int tap = kk / TAPSTR;
        int c2  = kk - tap * TAPSTR;
        if (c2 < 131) {
            int ic = (c2 < HID) ? (c2 + IN_CH) : (c2 - HID);
            v = conv_w[oc * KREAL + ic * 3 + tap];
        }
    } else if (kk == KBIAS) {
        v = conv_b[oc];
    }
    wh[idx] = (_Float16)v;
}

// ---------------------------------------------------------------------------
// Kernel 2: persistent ConvLSTM — one workgroup per batch element.
// 8 waves; wave w owns output-channel tiles [w*64, w*64+64).
// ---------------------------------------------------------------------------
__global__ __launch_bounds__(NTHREADS) void convlstm_persistent_kernel(
    const float* __restrict__ x,        // (B, T, 63)
    const _Float16* __restrict__ wh,    // (OC, KPAD) fp16 packed, tap-major
    const float* __restrict__ w1,       // (64, 128)
    const float* __restrict__ b1,       // (64,)
    const float* __restrict__ w2,       // (10, 64)
    const float* __restrict__ b2,       // (10,)
    float* __restrict__ out)            // (B, 10)
{
    __shared__ __align__(16) _Float16 sh_hT[NSENS * HID];   // h transposed: [s][hc], 5.25 KiB
    __shared__ float sh_c[HID * NSENS];                     // 10.5 KiB
    __shared__ __align__(32) unsigned char sh_u[OC * GSTRIDE * 4]; // 44 KiB union
    __shared__ float sh_feat[HID];
    __shared__ float sh_hid[64];

    _Float16* imcol = (_Float16*)sh_u;   // phase A: [NCOL][KPAD] fp16 (26.6 KiB)
    float*    gates = (float*)sh_u;      // phase B: [OC][GSTRIDE] f32 (44 KiB)

    const int tid  = threadIdx.x;
    const int b    = blockIdx.x;
    const int wave = tid >> 5;
    const int lane = tid & 31;
    const int lrow = lane & 15;
    const bool hiHalf = lane >= 16;

    // zero-init recurrent state
    for (int i = tid; i < NSENS * HID; i += NTHREADS) sh_hT[i] = (_Float16)0.0f;
    for (int i = tid; i < HID * NSENS; i += NTHREADS) sh_c[i] = 0.0f;

    // warm the weight image into L2 / WGP$ (global_prefetch_b8)
    {
        const char* wp = (const char*)wh;
        for (int off = tid * 128; off < OC * KPAD * 2; off += NTHREADS * 128)
            __builtin_prefetch(wp + off, 0, 1);
    }
    __syncthreads();

    const f32x8 vzero = {};
    const half8 hzero = {};

    #pragma unroll 1
    for (int t = 0; t < T_LEN; ++t) {
        const float* xt = x + ((size_t)b * T_LEN + t) * (NSENS * IN_CH);

        // ---- Phase 1a: vector zero-fill entire im2col ---------------------
        {
            half8* p = (half8*)imcol;
            for (int i = tid; i < (NCOL * KPAD) / 8; i += NTHREADS) p[i] = hzero;
        }
        __syncthreads();

        // ---- Phase 1b: block-copy h segments (16B chunks, both sides LDS) -
        // segment (col, tap): imcol[col][tap*136 + 0..127] = hT[col+tap-1][0..127]
        for (int w = tid; w < NSENS * 3 * 16; w += NTHREADS) {
            int seg   = w >> 4;          // 0..62
            int chunk = w & 15;          // 16 x half8 = 128 halves
            int col   = seg / 3;
            int tap   = seg - col * 3;
            int s     = col + tap - 1;
            if (s >= 0 && s < NSENS) {
                const half8* src = (const half8*)(sh_hT + s * HID);
                half8* dst = (half8*)(imcol + col * KPAD + tap * TAPSTR);
                dst[chunk] = src[chunk];
            }
        }
        // ---- Phase 1c: x channels + bias column ---------------------------
        for (int w = tid; w < NSENS * 10; w += NTHREADS) {
            int col = w / 10;
            int r   = w - col * 10;
            if (r < 9) {
                int tap = r / 3;
                int c   = r - tap * 3;
                int s   = col + tap - 1;
                if (s >= 0 && s < NSENS)
                    imcol[col * KPAD + tap * TAPSTR + HID + c] = (_Float16)xt[s * IN_CH + c];
            } else {
                imcol[col * KPAD + KBIAS] = (_Float16)1.0f;
            }
        }
        __syncthreads();

        // ---- Phase 2: GEMM via v_wmma_f32_16x16x32_f16 --------------------
        f32x8 acc[4][2];
        #pragma unroll
        for (int m = 0; m < 4; ++m) { acc[m][0] = vzero; acc[m][1] = vzero; }

        #pragma unroll
        for (int kt = 0; kt < KTILES; ++kt) {
            // B fragments from LDS: lane holds col=lane&15, K 0..15 (lo lanes)
            // or K 16..31 (hi lanes), contiguous halves -> 2x ds_load_b128.
            const int kbB = kt * 32 + (hiHalf ? 16 : 0);
            half16 bfrag[2];
            #pragma unroll
            for (int nt = 0; nt < 2; ++nt) {
                const _Float16* bp = imcol + (nt * 16 + lrow) * KPAD + kbB;
                half8 blo = *(const half8*)bp;
                half8 bhi = *(const half8*)(bp + 8);
                bfrag[nt] = __builtin_shufflevector(blo, bhi,
                    0,1,2,3,4,5,6,7,8,9,10,11,12,13,14,15);
            }
            // A fragments from global fp16 weights (L2-hot):
            // lane row = mtile*16 + (lane&15); lo lanes K{0..15},
            // hi lanes K{16..31} -> 2x global_load_b128.
            const int kbA = kt * 32 + (hiHalf ? 16 : 0);
            #pragma unroll
            for (int m = 0; m < 4; ++m) {
                int row = (wave * 4 + m) * 16 + lrow;
                const _Float16* ap = wh + (size_t)row * KPAD + kbA;
                half8 alo = *(const half8*)ap;
                half8 ahi = *(const half8*)(ap + 8);
                half16 afrag = __builtin_shufflevector(alo, ahi,
                    0,1,2,3,4,5,6,7,8,9,10,11,12,13,14,15);
                acc[m][0] = __builtin_amdgcn_wmma_f32_16x16x32_f16(
                    false, afrag, false, bfrag[0], (short)0, acc[m][0], false, false);
                acc[m][1] = __builtin_amdgcn_wmma_f32_16x16x32_f16(
                    false, afrag, false, bfrag[1], (short)0, acc[m][1], false, false);
            }
        }
        __syncthreads();   // all waves done reading imcol before gates overwrite

        // ---- Phase 3: spill accumulators to gate buffer -------------------
        // D layout: lane holds N=lane&15; VGPR r holds M = r + 8*(lane>=16).
        #pragma unroll
        for (int m = 0; m < 4; ++m) {
            int ocbase = (wave * 4 + m) * 16 + (hiHalf ? 8 : 0);
            #pragma unroll
            for (int nt = 0; nt < 2; ++nt) {
                int col = nt * 16 + lrow;
                if (col < NSENS) {
                    #pragma unroll
                    for (int r = 0; r < 8; ++r)
                        gates[(ocbase + r) * GSTRIDE + col] = acc[m][nt][r];
                }
            }
        }
        __syncthreads();

        // ---- Phase 4: LSTM cell update (bias already folded in) -----------
        const bool last = (t == T_LEN - 1);
        for (int idx = tid; idx < HID * NSENS; idx += NTHREADS) {
            int hc = idx / NSENS;
            int s  = idx - hc * NSENS;
            float gi = gates[(0 * HID + hc) * GSTRIDE + s];
            float gf = gates[(1 * HID + hc) * GSTRIDE + s];
            float go = gates[(2 * HID + hc) * GSTRIDE + s];
            float gg = gates[(3 * HID + hc) * GSTRIDE + s];
            float i  = 1.0f / (1.0f + __expf(-gi));
            float f  = 1.0f / (1.0f + __expf(-gf));
            float o  = 1.0f / (1.0f + __expf(-go));
            float g  = 1.0f - 2.0f / (__expf(2.0f * gg) + 1.0f);   // tanh
            float cn = f * sh_c[idx] + i * g;
            sh_c[idx] = cn;
            float tc = 1.0f - 2.0f / (__expf(2.0f * cn) + 1.0f);   // tanh
            float hn = o * tc;
            sh_hT[s * HID + hc] = (_Float16)hn;
            if (last) gates[hc * GSTRIDE + s] = hn;  // park f32 h for the head
        }
        __syncthreads();
    }

    // ---- Head: mean over sensors, then 128->64 relu MLP, then 64->10 ------
    if (tid < HID) {
        float sum = 0.0f;
        #pragma unroll
        for (int s = 0; s < NSENS; ++s) sum += gates[tid * GSTRIDE + s];
        sh_feat[tid] = sum * (1.0f / (float)NSENS);
    }
    __syncthreads();
    if (tid < 64) {
        float sum = b1[tid];
        #pragma unroll 8
        for (int k = 0; k < HID; ++k) sum += sh_feat[k] * w1[tid * HID + k];
        sh_hid[tid] = fmaxf(sum, 0.0f);
    }
    __syncthreads();
    if (tid < 10) {
        float sum = b2[tid];
        #pragma unroll 8
        for (int k = 0; k < 64; ++k) sum += sh_hid[k] * w2[tid * 64 + k];
        out[b * 10 + tid] = sum;
    }
}

// ---------------------------------------------------------------------------
extern "C" void kernel_launch(void* const* d_in, const int* in_sizes, int n_in,
                              void* d_out, int out_size, void* d_ws, size_t ws_size,
                              hipStream_t stream)
{
    const float* x      = (const float*)d_in[0];
    const float* conv_w = (const float*)d_in[1];
    const float* conv_b = (const float*)d_in[2];
    const float* w1     = (const float*)d_in[3];
    const float* b1     = (const float*)d_in[4];
    const float* w2     = (const float*)d_in[5];
    const float* b2     = (const float*)d_in[6];
    float* out = (float*)d_out;

    _Float16* wh = (_Float16*)d_ws;    // OC*KPAD fp16 = 416 KiB

    int n_pack = OC * KPAD;
    pack_weights_kernel<<<(n_pack + 255) / 256, 256, 0, stream>>>(conv_w, conv_b, wh);

    convlstm_persistent_kernel<<<BATCH, NTHREADS, 0, stream>>>(
        x, wh, w1, b1, w2, b2, out);
}
